// FusionM_21191368639133
// MI455X (gfx1250) — matrix operand order repaired and checked by reading, not compiled, not run on hardware
//
#include <hip/hip_runtime.h>

typedef __attribute__((ext_vector_type(16))) __bf16 v16bf;
typedef __attribute__((ext_vector_type(8)))  __bf16 v8bf;
typedef __attribute__((ext_vector_type(8)))  float  v8f;

#define B_    8
#define E_    768
#define C_    256
#define CI_   128
#define NS_   196     // 14*14
#define N_    3136    // 56*56
#define W2_   56

#define QPAD 136      // 64x136 bf16: 272 B pitch (17x16B), bank phase 68 DW
#define KPAD 136      // 32x136 per buffer
#define VPAD 40       // 128x40: 80 B pitch (5x16B), bank phase 20 DW
#define SPAD 36       // 16x36 f32: 144 B pitch (9x16B), bank phase 36 DW

// async global -> LDS, 16 B per lane (GLOBAL_LOAD_ASYNC_TO_LDS_B128, ASYNCcnt)
#define ASYNC_B128(ldsaddr, gptr, OFF)                                        \
    asm volatile("global_load_async_to_lds_b128 %0, %1, off offset:" OFF      \
                 :: "v"(ldsaddr), "v"(gptr) : "memory")

// build a 16-element bf16 fragment from two 16-byte chunks (LDS or global)
__device__ __forceinline__ v16bf frag_cat(const __bf16* p0, const __bf16* p1) {
    v8bf lo = *(const v8bf*)p0;
    v8bf hi = *(const v8bf*)p1;
    return __builtin_shufflevector(lo, hi, 0, 1, 2, 3, 4, 5, 6, 7, 8, 9, 10, 11, 12, 13, 14, 15);
}

// ---------------- Stage 0a: FCUUp conv E->C + BN + ReLU at 14x14 (0.3 GMAC, scalar) ----------------
__global__ void fcuup_kernel(const float* __restrict__ x_t, const float* __restrict__ Wp,
                             const float* __restrict__ bp, const float* __restrict__ g1,
                             const float* __restrict__ b1, const float* __restrict__ rm1,
                             const float* __restrict__ rv1, float* __restrict__ xr) {
    int idx = blockIdx.x * blockDim.x + threadIdx.x;           // [b][c][s]
    if (idx >= B_ * C_ * NS_) return;
    int s = idx % NS_;
    int c = (idx / NS_) % C_;
    int b = idx / (NS_ * C_);
    const float* xrow = x_t + ((size_t)b * (NS_ + 1) + 1 + s) * E_;
    const float* wrow = Wp + (size_t)c * E_;
    float acc = bp[c];
    for (int e = 0; e < E_; ++e) acc += wrow[e] * xrow[e];
    float inv = g1[c] * rsqrtf(rv1[c] + 1e-6f);
    acc = acc * inv + (b1[c] - rm1[c] * inv);
    xr[idx] = fmaxf(acc, 0.0f);
}

// ---------------- Stage 0b: phi at small resolution, stored [b][s][ci] bf16 ----------------
__global__ void phi_kernel(const float* __restrict__ xr, const float* __restrict__ Wph,
                           const float* __restrict__ bph, __bf16* __restrict__ phi_q) {
    int idx = blockIdx.x * blockDim.x + threadIdx.x;           // [b][s][ci]
    if (idx >= B_ * NS_ * CI_) return;
    int ci = idx % CI_;
    int s  = (idx / CI_) % NS_;
    int b  = idx / (CI_ * NS_);
    const float* w = Wph + (size_t)ci * C_;
    const float* x = xr + (size_t)b * C_ * NS_ + s;
    float acc = bph[ci];
    for (int c = 0; c < C_; ++c) acc += w[c] * x[(size_t)c * NS_];
    phi_q[idx] = (__bf16)acc;
}

// ---------------- conversions: x_cnn -> xT bf16 [b][n][c]; weights f32 -> bf16 ----------------
__global__ void xT_kernel(const float* __restrict__ x_cnn, __bf16* __restrict__ xT) {
    int idx = blockIdx.x * blockDim.x + threadIdx.x;           // [b][n][c]
    if (idx >= B_ * N_ * C_) return;
    int c = idx % C_;
    int n = (idx / C_) % N_;
    int b = idx / (C_ * N_);
    xT[idx] = (__bf16)x_cnn[((size_t)b * C_ + c) * N_ + n];
}

__global__ void cvt_bf16_kernel(const float* __restrict__ src, __bf16* __restrict__ dst, int n) {
    int i = blockIdx.x * blockDim.x + threadIdx.x;
    if (i < n) dst[i] = (__bf16)src[i];
}

// ---------------- fold BN2 into per-channel scale/bias ----------------
__global__ void bnprep_kernel(const float* __restrict__ bz, const float* __restrict__ g2,
                              const float* __restrict__ b2, const float* __restrict__ rm2,
                              const float* __restrict__ rv2, float* __restrict__ inv2,
                              float* __restrict__ bias2) {
    int c = blockIdx.x * blockDim.x + threadIdx.x;
    if (c >= C_) return;
    float inv = g2[c] * rsqrtf(rv2[c] + 1e-5f);
    inv2[c] = inv;
    bias2[c] = bz[c] * inv + b2[c] - rm2[c] * inv;
}

// ---------------- Stage 1a: theta = W[128x256] * X, output [b][n][ci] (M = n orientation) ----
// grid (49, 8), block 128; wave handles 16 n-rows x 128 ci-cols, K=256; frags straight from global
__global__ void __launch_bounds__(128) theta_wmma_kernel(const __bf16* __restrict__ xT,
                                                         const __bf16* __restrict__ Wbf,
                                                         const float* __restrict__ bias,
                                                         __bf16* __restrict__ thT) {
    const int b    = blockIdx.y;
    const int tid  = threadIdx.x;
    const int wave = tid >> 5, lane = tid & 31, lh = lane >> 4, col = lane & 15;
    const int nbase = blockIdx.x * 64 + wave * 16;
    const __bf16* xrow = xT + ((size_t)b * N_ + nbase + col) * C_;   // A row (M = n)
    v8f acc[8];
#pragma unroll
    for (int ct = 0; ct < 8; ++ct)
#pragma unroll
        for (int j = 0; j < 8; ++j) acc[ct][j] = 0.0f;

    for (int ch = 0; ch < 8; ++ch) {
        v16bf af = frag_cat(xrow + ch * 32 + 8 * lh, xrow + ch * 32 + 16 + 8 * lh);
#pragma unroll
        for (int ct = 0; ct < 8; ++ct) {
            const __bf16* wrow = Wbf + (size_t)(ct * 16 + col) * C_ + ch * 32 + 8 * lh;
            v16bf bf = frag_cat(wrow, wrow + 16);
            acc[ct] = __builtin_amdgcn_wmma_f32_16x16x32_bf16(false, af, false, bf, (short)0, acc[ct], false, false);
        }
    }
#pragma unroll
    for (int ct = 0; ct < 8; ++ct) {
        float bb = bias[ct * 16 + col];
#pragma unroll
        for (int j = 0; j < 8; ++j) {
            int n = nbase + j + 8 * lh;
            thT[((size_t)b * N_ + n) * CI_ + ct * 16 + col] = (__bf16)(acc[ct][j] + bb);
        }
    }
}

// ---------------- Stage 1b: g = W[128x256] * X, output [b][ci][n] (M = ci orientation) ----
__global__ void __launch_bounds__(128) g_wmma_kernel(const __bf16* __restrict__ xT,
                                                     const __bf16* __restrict__ Wbf,
                                                     const float* __restrict__ bias,
                                                     __bf16* __restrict__ gN) {
    const int b    = blockIdx.y;
    const int tid  = threadIdx.x;
    const int wave = tid >> 5, lane = tid & 31, lh = lane >> 4, col = lane & 15;
    const int nbase = blockIdx.x * 64 + wave * 16;
    const __bf16* xrow = xT + ((size_t)b * N_ + nbase + col) * C_;   // B col (n)
    v8f acc[8];
#pragma unroll
    for (int ot = 0; ot < 8; ++ot)
#pragma unroll
        for (int j = 0; j < 8; ++j) acc[ot][j] = 0.0f;

    for (int ch = 0; ch < 8; ++ch) {
        v16bf bf = frag_cat(xrow + ch * 32 + 8 * lh, xrow + ch * 32 + 16 + 8 * lh);
#pragma unroll
        for (int ot = 0; ot < 8; ++ot) {
            const __bf16* wrow = Wbf + (size_t)(ot * 16 + col) * C_ + ch * 32;
            v16bf af = frag_cat(wrow + 8 * lh, wrow + 16 + 8 * lh);
            acc[ot] = __builtin_amdgcn_wmma_f32_16x16x32_bf16(false, af, false, bf, (short)0, acc[ot], false, false);
        }
    }
#pragma unroll
    for (int ot = 0; ot < 8; ++ot)
#pragma unroll
        for (int j = 0; j < 8; ++j) {
            int o = ot * 16 + j + 8 * lh;
            gN[((size_t)b * CI_ + o) * N_ + nbase + col] = (__bf16)(acc[ot][j] + bias[o]);
        }
}

// ---------------- Stage 2: fused flash attention, bf16 WMMA + f32 accumulate ----------------
// grid (49, 8), block 128 = 4 waves; wave w owns query rows [mblk*64 + w*16, +16)
__global__ void __launch_bounds__(128) attn_kernel(const __bf16* __restrict__ phi_q,
                                                   const __bf16* __restrict__ thT,
                                                   const __bf16* __restrict__ gN,
                                                   __bf16* __restrict__ y) {
    __shared__ __bf16 Qs[64][QPAD];        // queries [row][ci]          17,408 B
    __shared__ __bf16 Kt[2][32][KPAD];     // keys    [buf][nloc][ci]    17,408 B
    __shared__ __bf16 Vt[2][CI_][VPAD];    // values  [buf][ci][nloc]    20,480 B
    __shared__ float  Sst[4][16][SPAD];    // per-wave raw S staging      9,216 B

    const int mblk = blockIdx.x;
    const int b    = blockIdx.y;
    const int tid  = threadIdx.x;
    const int wave = tid >> 5;
    const int lane = tid & 31;
    const int lh   = lane >> 4;    // lane half
    const int col  = lane & 15;

    const __bf16* thb = thT + (size_t)b * N_ * CI_;
    const __bf16* gb  = gN  + (size_t)b * CI_ * N_;
    const int kn  = tid >> 2;          // key row 0..31
    const int kci = (tid & 3) * 32;    // 32-channel quarter

    // ---- cooperative Q load (query m -> small pixel via nearest-upsample map) ----
    {
        const int r   = tid >> 1;
        const int ci0 = (tid & 1) * 64;
        const int m   = mblk * 64 + r;
        const int h2 = m / W2_, w2 = m % W2_;
        const int sm = (h2 >> 2) * 14 + (w2 >> 2);
        const uint4* s4 = (const uint4*)(phi_q + ((size_t)b * NS_ + sm) * CI_ + ci0);
        uint4* d4 = (uint4*)(&Qs[r][ci0]);
#pragma unroll
        for (int i = 0; i < 8; ++i) d4[i] = s4[i];
    }

    // ---- issue async loads for tile 0 into buffer 0 (no VGPR staging) ----
    {
        unsigned lk = (unsigned)(size_t)&Kt[0][kn][kci];
        const __bf16* gk = thb + (size_t)kn * CI_ + kci;
        ASYNC_B128(lk, gk, "0");  ASYNC_B128(lk, gk, "16");
        ASYNC_B128(lk, gk, "32"); ASYNC_B128(lk, gk, "48");
        unsigned lv = (unsigned)(size_t)&Vt[0][tid][0];
        const __bf16* gv = gb + (size_t)tid * N_;
        ASYNC_B128(lv, gv, "0");  ASYNC_B128(lv, gv, "16");
        ASYNC_B128(lv, gv, "32"); ASYNC_B128(lv, gv, "48");
    }
    __syncthreads();

    // ---- Q A-fragments: two contiguous 16B chunks per (chunk, lane) ----
    const int qrow = wave * 16 + col;
    v16bf qa[4];
#pragma unroll
    for (int ch = 0; ch < 4; ++ch)
        qa[ch] = frag_cat(&Qs[qrow][ch * 32 + 8 * lh], &Qs[qrow][ch * 32 + 16 + 8 * lh]);

    v8f o[8];
#pragma unroll
    for (int cc = 0; cc < 8; ++cc)
#pragma unroll
        for (int j = 0; j < 8; ++j) o[cc][j] = 0.0f;

    // flash state in the row-owner domain: lane `col` owns row `col` of the wave tile
    float m_run = -1e30f, l_run = 0.0f;

    for (int nblk = 0, it = 0; nblk < N_; nblk += 32, ++it) {
        const int buf = it & 1;
        asm volatile("s_wait_asynccnt 0x0" ::: "memory");   // my tile loads landed in LDS
        __syncthreads();                                    // everyone's landed; prev reads done

        if (nblk + 32 < N_) {   // async-fetch next tile into the other buffer
            unsigned lk = (unsigned)(size_t)&Kt[buf ^ 1][kn][kci];
            const __bf16* gk = thb + (size_t)(nblk + 32 + kn) * CI_ + kci;
            ASYNC_B128(lk, gk, "0");  ASYNC_B128(lk, gk, "16");
            ASYNC_B128(lk, gk, "32"); ASYNC_B128(lk, gk, "48");
            unsigned lv = (unsigned)(size_t)&Vt[buf ^ 1][tid][0];
            const __bf16* gv = gb + (size_t)tid * N_ + nblk + 32;
            ASYNC_B128(lv, gv, "0");  ASYNC_B128(lv, gv, "16");
            ASYNC_B128(lv, gv, "32"); ASYNC_B128(lv, gv, "48");
        }
        if (nblk + 64 < N_) {   // global_prefetch_b8 two tiles ahead
            __builtin_prefetch(thb + (size_t)(nblk + 64 + kn) * CI_ + kci, 0, 1);
            __builtin_prefetch(gb + (size_t)tid * N_ + nblk + 64, 0, 1);
        }

        // ---- S = Q * K^T (16 x 32), 8 WMMAs; B-frags contiguous ds_load_b128 pairs ----
        v8f s0, s1;
#pragma unroll
        for (int j = 0; j < 8; ++j) { s0[j] = 0.0f; s1[j] = 0.0f; }
#pragma unroll
        for (int ch = 0; ch < 4; ++ch) {
            v16bf kb0 = frag_cat(&Kt[buf][col][ch * 32 + 8 * lh],
                                 &Kt[buf][col][ch * 32 + 16 + 8 * lh]);
            v16bf kb1 = frag_cat(&Kt[buf][16 + col][ch * 32 + 8 * lh],
                                 &Kt[buf][16 + col][ch * 32 + 16 + 8 * lh]);
            s0 = __builtin_amdgcn_wmma_f32_16x16x32_bf16(false, qa[ch], false, kb0, (short)0, s0, false, false);
            s1 = __builtin_amdgcn_wmma_f32_16x16x32_bf16(false, qa[ch], false, kb1, (short)0, s1, false, false);
        }

        // ---- stage raw S (f32) to wave-private LDS, row-major ----
#pragma unroll
        for (int j = 0; j < 8; ++j) {
            Sst[wave][j + 8 * lh][col]      = s0[j];
            Sst[wave][j + 8 * lh][col + 16] = s1[j];
        }
        asm volatile("s_wait_dscnt 0x0" ::: "memory");   // same-wave LDS ordering

        // ---- row-owner softmax: lane reads its row's 16 columns (A-layout order) ----
        float sv[16];
        {
            const float* srow = &Sst[wave][col][0];
            *(float4*)&sv[0]  = *(const float4*)(srow + 8 * lh);
            *(float4*)&sv[4]  = *(const float4*)(srow + 8 * lh + 4);
            *(float4*)&sv[8]  = *(const float4*)(srow + 16 + 8 * lh);
            *(float4*)&sv[12] = *(const float4*)(srow + 16 + 8 * lh + 4);
        }
        float tmax = sv[0];
#pragma unroll
        for (int t = 1; t < 16; ++t) tmax = fmaxf(tmax, sv[t]);
        tmax = fmaxf(tmax, __shfl_xor(tmax, 16, 32));        // combine the two halves
        const float nm = fmaxf(m_run, tmax);
        const float sc = __expf(m_run - nm);

        v16bf pa;        // P fragment built directly in A-layout
        float rsum = 0.0f;
#pragma unroll
        for (int t = 0; t < 16; ++t) {
            float p = __expf(sv[t] - nm);
            rsum += p;
            pa[t] = (__bf16)p;
        }
        rsum += __shfl_xor(rsum, 16, 32);
        l_run = l_run * sc + rsum;
        m_run = nm;

        // broadcast per-row rescale into the C-layout domain (8 bpermutes)
#pragma unroll
        for (int j = 0; j < 8; ++j) {
            float scj = __shfl(sc, j + 8 * lh, 32);
#pragma unroll
            for (int cc = 0; cc < 8; ++cc) o[cc][j] *= scj;
        }

        // ---- O += P * V (16 x 128), 8 WMMAs ----
#pragma unroll
        for (int cc = 0; cc < 8; ++cc) {
            v16bf vb = frag_cat(&Vt[buf][cc * 16 + col][8 * lh],
                                &Vt[buf][cc * 16 + col][16 + 8 * lh]);
            o[cc] = __builtin_amdgcn_wmma_f32_16x16x32_bf16(false, pa, false, vb, (short)0, o[cc], false, false);
        }
    }

    // ---- epilogue: normalize rows, write ybf[b][m][ci] bf16 (feeds Wz WMMA stage) ----
    float lrj[8];
#pragma unroll
    for (int j = 0; j < 8; ++j) lrj[j] = __shfl(l_run, j + 8 * lh, 32);
#pragma unroll
    for (int cc = 0; cc < 8; ++cc) {
#pragma unroll
        for (int j = 0; j < 8; ++j) {
            int row = j + 8 * lh;
            int m = mblk * 64 + wave * 16 + row;
            int ci = cc * 16 + col;
            y[((size_t)b * N_ + m) * CI_ + ci] = (__bf16)(o[cc][j] / lrj[j]);
        }
    }
}

// ---------------- Stage 3: W_z [256x128] * y + BN + residual, output [b][c][n] ----------------
// grid (98, 8), block 128; wave -> (n-subtile, channel-half); K=128
__global__ void __launch_bounds__(128) out_wmma_kernel(const __bf16* __restrict__ ybf,
                                                       const __bf16* __restrict__ Wzbf,
                                                       const float* __restrict__ inv2,
                                                       const float* __restrict__ bias2,
                                                       const float* __restrict__ x_cnn,
                                                       float* __restrict__ out) {
    const int b    = blockIdx.y;
    const int tid  = threadIdx.x;
    const int wave = tid >> 5, lane = tid & 31, lh = lane >> 4, col = lane & 15;
    const int nsub = wave >> 1, ohalf = wave & 1;
    const int nbase = blockIdx.x * 32 + nsub * 16;
    const __bf16* yrow = ybf + ((size_t)b * N_ + nbase + col) * CI_;   // B col (n)
    v8f acc[8];
#pragma unroll
    for (int ot = 0; ot < 8; ++ot)
#pragma unroll
        for (int j = 0; j < 8; ++j) acc[ot][j] = 0.0f;

    for (int ch = 0; ch < 4; ++ch) {
        v16bf bf = frag_cat(yrow + ch * 32 + 8 * lh, yrow + ch * 32 + 16 + 8 * lh);
#pragma unroll
        for (int ot = 0; ot < 8; ++ot) {
            const __bf16* wrow = Wzbf + (size_t)(ohalf * 128 + ot * 16 + col) * CI_ + ch * 32;
            v16bf af = frag_cat(wrow + 8 * lh, wrow + 16 + 8 * lh);
            acc[ot] = __builtin_amdgcn_wmma_f32_16x16x32_bf16(false, af, false, bf, (short)0, acc[ot], false, false);
        }
    }
#pragma unroll
    for (int ot = 0; ot < 8; ++ot)
#pragma unroll
        for (int j = 0; j < 8; ++j) {
            int c = ohalf * 128 + ot * 16 + j + 8 * lh;
            size_t oi = ((size_t)b * C_ + c) * N_ + nbase + col;
            out[oi] = acc[ot][j] * inv2[c] + bias2[c] + x_cnn[oi];
        }
}

extern "C" void kernel_launch(void* const* d_in, const int* in_sizes, int n_in,
                              void* d_out, int out_size, void* d_ws, size_t ws_size,
                              hipStream_t stream) {
    (void)in_sizes; (void)n_in; (void)out_size; (void)ws_size;
    const float* x_t   = (const float*)d_in[0];
    const float* x_cnn = (const float*)d_in[1];
    const float* Wp  = (const float*)d_in[2];
    const float* bp  = (const float*)d_in[3];
    const float* g1  = (const float*)d_in[4];
    const float* b1  = (const float*)d_in[5];
    const float* rm1 = (const float*)d_in[6];
    const float* rv1 = (const float*)d_in[7];
    const float* Wg  = (const float*)d_in[8];
    const float* bg  = (const float*)d_in[9];
    const float* Wth = (const float*)d_in[10];
    const float* bth = (const float*)d_in[11];
    const float* Wph = (const float*)d_in[12];
    const float* bph = (const float*)d_in[13];
    const float* Wz  = (const float*)d_in[14];
    const float* bz  = (const float*)d_in[15];
    const float* g2  = (const float*)d_in[16];
    const float* b2  = (const float*)d_in[17];
    const float* rm2 = (const float*)d_in[18];
    const float* rv2 = (const float*)d_in[19];

    // workspace layout (256B-aligned offsets)
    char* ws = (char*)d_ws;
    float*  xr    = (float*)(ws + 0);            // B*C*196 f32      = 1,605,632 B
    __bf16* phi_q = (__bf16*)(ws + 1605632);     // B*196*128 bf16   =   401,408 B
    __bf16* thT   = (__bf16*)(ws + 2007040);     // B*3136*128 bf16  = 6,422,528 B
    __bf16* gNb   = (__bf16*)(ws + 8429568);     // B*128*3136 bf16  = 6,422,528 B
    __bf16* ybf   = (__bf16*)(ws + 14852096);    // B*3136*128 bf16  = 6,422,528 B
    __bf16* xTb   = (__bf16*)(ws + 21274624);    // B*3136*256 bf16  = 12,845,056 B
    __bf16* Wthbf = (__bf16*)(ws + 34119680);    // 128*256 bf16     =    65,536 B
    __bf16* Wgbf  = (__bf16*)(ws + 34185216);    // 128*256 bf16     =    65,536 B
    __bf16* Wzbf  = (__bf16*)(ws + 34250752);    // 256*128 bf16     =    65,536 B
    float*  inv2  = (float*)(ws + 34316288);     // 256 f32
    float*  bias2 = (float*)(ws + 34317312);     // 256 f32
    float* out = (float*)d_out;

    fcuup_kernel<<<(B_ * C_ * NS_ + 255) / 256, 256, 0, stream>>>(x_t, Wp, bp, g1, b1, rm1, rv1, xr);
    phi_kernel<<<(B_ * NS_ * CI_ + 255) / 256, 256, 0, stream>>>(xr, Wph, bph, phi_q);
    xT_kernel<<<(B_ * N_ * C_ + 255) / 256, 256, 0, stream>>>(x_cnn, xTb);
    cvt_bf16_kernel<<<(CI_ * C_ + 255) / 256, 256, 0, stream>>>(Wth, Wthbf, CI_ * C_);
    cvt_bf16_kernel<<<(CI_ * C_ + 255) / 256, 256, 0, stream>>>(Wg, Wgbf, CI_ * C_);
    cvt_bf16_kernel<<<(C_ * CI_ + 255) / 256, 256, 0, stream>>>(Wz, Wzbf, C_ * CI_);
    bnprep_kernel<<<1, 256, 0, stream>>>(bz, g2, b2, rm2, rv2, inv2, bias2);
    theta_wmma_kernel<<<dim3(N_ / 64, B_), 128, 0, stream>>>(xTb, Wthbf, bth, thT);
    g_wmma_kernel<<<dim3(N_ / 64, B_), 128, 0, stream>>>(xTb, Wgbf, bg, gNb);
    attn_kernel<<<dim3(N_ / 64, B_), 128, 0, stream>>>(phi_q, thT, gNb, ybf);
    out_wmma_kernel<<<dim3(N_ / 32, B_), 128, 0, stream>>>(ybf, Wzbf, inv2, bias2, x_cnn, out);
}
